// KiperwasserDependencyParser_40407052321078
// MI455X (gfx1250) — compile-verified, matrix operand order checked by reading, not run on hardware
//
#include <hip/hip_runtime.h>
#include <hip/hip_bf16.h>

typedef __attribute__((ext_vector_type(16))) __bf16 v16bf;
typedef __attribute__((ext_vector_type(8)))  float  v8f;

#define TOK 1024      // sequence length N
#define HID 125       // LSTM hidden per direction
#define GATES 500     // 4*HID
#define PROJ 100      // projection dim

__device__ __forceinline__ float fast_tanh(float x) {
    float r;
    asm volatile("v_tanh_f32 %0, %1" : "=v"(r) : "v"(x));
    return r;
}
__device__ __forceinline__ float sigm(float x) {
    return 1.0f / (1.0f + __expf(-x));
}

// Load one 16x32 bf16 fragment row-chunk: two contiguous 16B chunks
// (K = kb..kb+7 and kb+16..kb+23) per the documented wave32 16-bit layout.
__device__ __forceinline__ v16bf load_frag(const __bf16* __restrict__ p) {
    v16bf f;
    *reinterpret_cast<uint4*>(&f)     = *reinterpret_cast<const uint4*>(p);
    *(reinterpret_cast<uint4*>(&f)+1) = *reinterpret_cast<const uint4*>(p + 16);
    return f;
}

// ---------------- embedding gather (f32 -> bf16, padded lda=128) -------------
__global__ void embed_kernel(const int* __restrict__ wi, const int* __restrict__ pi,
                             const float* __restrict__ wemb, const float* __restrict__ pemb,
                             __bf16* __restrict__ e) {
    int idx = blockIdx.x * blockDim.x + threadIdx.x;       // over TOK*128
    if (idx >= TOK * 128) return;
    int i = idx >> 7, k = idx & 127;
    float v = 0.f;
    if (k < 100)       v = wemb[(size_t)wi[i] * 100 + k];
    else if (k < 125)  v = pemb[(size_t)pi[i] * 25 + (k - 100)];
    e[idx] = (__bf16)v;
}

// ---------------- f32 -> bf16 convert with zero padding ----------------------
// dst is (nrows_pad x drow); src rows are srow-strided starting at scol_off.
__global__ void cvt_pad_kernel(const float* __restrict__ src, int srow, int scol_off,
                               __bf16* __restrict__ dst, int drow,
                               int nrows_src, int ncols_src, int total) {
    int idx = blockIdx.x * blockDim.x + threadIdx.x;
    if (idx >= total) return;
    int r = idx / drow, c = idx - r * drow;
    float v = (r < nrows_src && c < ncols_src) ? src[(size_t)r * srow + scol_off + c] : 0.f;
    dst[idx] = (__bf16)v;
}

__global__ void fill_zero_bf16(__bf16* __restrict__ p, int n) {
    int i = blockIdx.x * blockDim.x + threadIdx.x;
    if (i < n) p[i] = (__bf16)0.f;
}

// ---------------- WMMA GEMM (LDS-free): C = A @ B^T --------------------------
// A: (M x K) bf16, lda mult of 8, K mult of 32, M mult of 64.
// B: (Npad x K) bf16 row-major, Npad mult of 16, zero padded rows.
// block = 128 (4 waves); wave w -> M tile blockIdx.y*64 + w*16; grid.x over Npad/16.
__global__ void gemm_bf16_wmma(const __bf16* __restrict__ A, int lda,
                               const __bf16* __restrict__ B, int ldb,
                               float* __restrict__ C, int ldc,
                               int Nstore, int K) {
    const int wave = threadIdx.x >> 5;
    const int lane = threadIdx.x & 31;
    const int n0   = blockIdx.x * 16;
    const int m0w  = blockIdx.y * 64 + wave * 16;
    const int kb   = (lane >> 4) * 8;         // K base for this lane half

    const __bf16* Ap = A + (size_t)(m0w + (lane & 15)) * lda + kb;
    const __bf16* Bp = B + (size_t)(n0  + (lane & 15)) * ldb + kb;

    v8f acc = {};
    for (int k0 = 0; k0 < K; k0 += 32) {
        v16bf a = load_frag(Ap + k0);
        v16bf b = load_frag(Bp + k0);
        acc = __builtin_amdgcn_wmma_f32_16x16x32_bf16(false, a, false, b,
                                                      (short)0, acc, false, false);
    }

    const int col = n0 + (lane & 15);
    if (col < Nstore) {
#pragma unroll
        for (int r = 0; r < 8; ++r) {
            int row = m0w + r + ((lane >> 4) << 3);
            C[(size_t)row * ldc + col] = acc[r];
        }
    }
}

// ---------------- sequential BiLSTM direction kernel -------------------------
// grid.x = 2 (dir 0 fwd, dir 1 bwd), block = 512 threads.
// Whh (2 x GATES x HID, f32) is parked in LDS (250KB of the 320KB/WGP budget).
__global__ void lstm_dir_kernel(const float* __restrict__ gx,
                                const float* __restrict__ whh,
                                const float* __restrict__ bih,
                                const float* __restrict__ bhh,
                                __bf16* __restrict__ hcat) {
    extern __shared__ float smem[];
    float* swhh = smem;                 // GATES*HID = 62500
    float* sh   = smem + GATES * HID;   // HID
    float* sg   = sh + HID;             // GATES
    float* sb   = sg + GATES;           // GATES

    const int dir  = blockIdx.x;
    const int rev  = dir;
    const int coff = dir * HID;
    const float* gxd  = gx  + (size_t)dir * TOK * GATES;
    const float* whhd = whh + (size_t)dir * GATES * HID;

    const int tid = threadIdx.x;
    for (int idx = tid; idx < GATES * HID; idx += 512) swhh[idx] = whhd[idx];
    if (tid < GATES) sb[tid] = bih[dir * GATES + tid] + bhh[dir * GATES + tid];
    if (tid < HID)   sh[tid] = 0.f;
    float c = 0.f;
    __syncthreads();

    for (int step = 0; step < TOK; ++step) {
        const int t = rev ? (TOK - 1 - step) : step;
        if (tid < GATES) {
            const float* wr = swhh + tid * HID;
            float acc = 0.f;
#pragma unroll 5
            for (int k = 0; k < HID; ++k) acc += sh[k] * wr[k];
            sg[tid] = acc + gxd[(size_t)t * GATES + tid] + sb[tid];
        }
        __syncthreads();
        if (tid < HID) {
            float gi = sg[tid];
            float gf = sg[HID + tid];
            float gc = sg[2 * HID + tid];
            float go = sg[3 * HID + tid];
            c = sigm(gf) * c + sigm(gi) * fast_tanh(gc);
            float h = sigm(go) * fast_tanh(c);
            sh[tid] = h;
            hcat[(size_t)t * 256 + coff + tid] = (__bf16)h;   // padded ld=256
        }
        __syncthreads();
    }
}

// ---------------- fused biaffine scoring: sT[j][i] = score(i,j) --------------
// 32x32 block tile, 2x2 micro-tile per thread -> ~1 LDS read per v_tanh.
// b1 folded into the staged ph tile.
__global__ void score_kernel(const float* __restrict__ ph, const float* __restrict__ pm,
                             const float* __restrict__ b1, const float* __restrict__ w2,
                             const float* __restrict__ b2, float* __restrict__ sT) {
    __shared__ float sph[32][PROJ];
    __shared__ float spm[32][PROJ];
    __shared__ float sw2[PROJ];

    const int i0 = blockIdx.x * 32;
    const int j0 = blockIdx.y * 32;
    const int tx = threadIdx.x, ty = threadIdx.y;    // 16 x 16
    const int tid = ty * 16 + tx;

    for (int idx = tid; idx < 32 * PROJ; idx += 256) {
        int r = idx / PROJ, k = idx - r * PROJ;
        sph[r][k] = ph[(size_t)(i0 + r) * PROJ + k] + b1[k];
        spm[r][k] = pm[(size_t)(j0 + r) * PROJ + k];
    }
    if (tid < PROJ) sw2[tid] = w2[tid];
    __syncthreads();

    const float bias2 = b2[0];
    float a00 = bias2, a01 = bias2, a10 = bias2, a11 = bias2;
#pragma unroll 4
    for (int k = 0; k < PROJ; ++k) {
        float p0 = sph[tx][k],      p1 = sph[tx + 16][k];
        float q0 = spm[ty][k],      q1 = spm[ty + 16][k];
        float w  = sw2[k];
        a00 += fast_tanh(p0 + q0) * w;
        a01 += fast_tanh(p0 + q1) * w;
        a10 += fast_tanh(p1 + q0) * w;
        a11 += fast_tanh(p1 + q1) * w;
    }
    sT[(size_t)(j0 + ty)      * TOK + (i0 + tx)]      = a00;
    sT[(size_t)(j0 + ty + 16) * TOK + (i0 + tx)]      = a01;
    sT[(size_t)(j0 + ty)      * TOK + (i0 + tx + 16)] = a10;
    sT[(size_t)(j0 + ty + 16) * TOK + (i0 + tx + 16)] = a11;
}

// ---------------- per-column log_softmax (axis 0) ----------------------------
__global__ void colsoftmax_kernel(const float* __restrict__ sT, float* __restrict__ out) {
    __shared__ float red[256];
    const int j = blockIdx.x;
    const int tid = threadIdx.x;
    const float* row = sT + (size_t)j * TOK;

    float m = -1e30f;
    for (int i = tid; i < TOK; i += 256) m = fmaxf(m, row[i]);
    red[tid] = m; __syncthreads();
    for (int s = 128; s > 0; s >>= 1) {
        if (tid < s) red[tid] = fmaxf(red[tid], red[tid + s]);
        __syncthreads();
    }
    m = red[0]; __syncthreads();

    float sum = 0.f;
    for (int i = tid; i < TOK; i += 256) sum += __expf(row[i] - m);
    red[tid] = sum; __syncthreads();
    for (int s = 128; s > 0; s >>= 1) {
        if (tid < s) red[tid] += red[tid + s];
        __syncthreads();
    }
    const float lse = __logf(red[0]);

    for (int i = tid; i < TOK; i += 256)
        out[(size_t)i * TOK + j] = row[i] - m - lse;
}

// ============================================================================
extern "C" void kernel_launch(void* const* d_in, const int* in_sizes, int n_in,
                              void* d_out, int out_size, void* d_ws, size_t ws_size,
                              hipStream_t stream) {
    const int*   word_idx = (const int*)  d_in[0];
    const int*   pos_idx  = (const int*)  d_in[1];
    const float* word_emb = (const float*)d_in[2];
    const float* pos_emb  = (const float*)d_in[3];
    const float* Wih0     = (const float*)d_in[4];
    const float* Whh0     = (const float*)d_in[5];
    const float* bih0     = (const float*)d_in[6];
    const float* bhh0     = (const float*)d_in[7];
    const float* Wih1     = (const float*)d_in[8];
    const float* Whh1     = (const float*)d_in[9];
    const float* bih1     = (const float*)d_in[10];
    const float* bhh1     = (const float*)d_in[11];
    const float* W1       = (const float*)d_in[12];
    const float* b1       = (const float*)d_in[13];
    const float* W2       = (const float*)d_in[14];
    const float* b2       = (const float*)d_in[15];
    float* out = (float*)d_out;

    // ---- workspace layout (all buffers 512B aligned => 16B vector loads ok) ----
    char* ws = (char*)d_ws;
    size_t off = 0;
    auto alloc = [&](size_t bytes) { size_t p = off; off = (off + bytes + 511) & ~(size_t)511; return p; };

    __bf16* e_bf16 = (__bf16*)(ws + alloc((size_t)TOK * 128 * 2));       // 1024x128
    __bf16* wih0p  = (__bf16*)(ws + alloc((size_t)2 * 512 * 128 * 2));   // 2 x (512x128)
    __bf16* wih1p  = (__bf16*)(ws + alloc((size_t)2 * 512 * 256 * 2));   // 2 x (512x256)
    __bf16* whp    = (__bf16*)(ws + alloc((size_t)112 * 256 * 2));       // Wh padded
    __bf16* wmp    = (__bf16*)(ws + alloc((size_t)112 * 256 * 2));       // Wm padded
    float*  gx0    = (float*) (ws + alloc((size_t)2 * TOK * GATES * 4));
    float*  gx1    = (float*) (ws + alloc((size_t)2 * TOK * GATES * 4));
    __bf16* h0cat  = (__bf16*)(ws + alloc((size_t)TOK * 256 * 2));       // padded ld=256
    __bf16* h1cat  = (__bf16*)(ws + alloc((size_t)TOK * 256 * 2));
    float*  ph     = (float*) (ws + alloc((size_t)TOK * PROJ * 4));
    float*  pm     = (float*) (ws + alloc((size_t)TOK * PROJ * 4));
    float*  sT     = (float*) (ws + alloc((size_t)TOK * TOK * 4));

    // 1) embeddings -> bf16 (zero-padded cols 125..127)
    embed_kernel<<<(TOK * 128 + 255) / 256, 256, 0, stream>>>(word_idx, pos_idx, word_emb, pos_emb, e_bf16);

    // 2) weight conversions into zero-padded tiles
    for (int d = 0; d < 2; ++d) {
        cvt_pad_kernel<<<(512 * 128 + 255) / 256, 256, 0, stream>>>(
            Wih0 + (size_t)d * GATES * 125, 125, 0, wih0p + (size_t)d * 512 * 128, 128, GATES, 125, 512 * 128);
        cvt_pad_kernel<<<(512 * 256 + 255) / 256, 256, 0, stream>>>(
            Wih1 + (size_t)d * GATES * 250, 250, 0, wih1p + (size_t)d * 512 * 256, 256, GATES, 250, 512 * 256);
    }
    cvt_pad_kernel<<<(112 * 256 + 255) / 256, 256, 0, stream>>>(W1, 500, 0,   whp, 256, PROJ, 250, 112 * 256);
    cvt_pad_kernel<<<(112 * 256 + 255) / 256, 256, 0, stream>>>(W1, 500, 250, wmp, 256, PROJ, 250, 112 * 256);

    // zero-fill padded hidden buffers (cols 250..255 must be 0 for next GEMM)
    fill_zero_bf16<<<(TOK * 256 + 255) / 256, 256, 0, stream>>>(h0cat, TOK * 256);
    fill_zero_bf16<<<(TOK * 256 + 255) / 256, 256, 0, stream>>>(h1cat, TOK * 256);

    const size_t lstm_smem = (size_t)(GATES * HID + HID + 2 * GATES) * sizeof(float); // ~254.5 KB

    // 3) layer-0 input GEMMs: gx0[d] = e @ Wih0[d].T (K padded 125->128)
    for (int d = 0; d < 2; ++d)
        gemm_bf16_wmma<<<dim3(32, TOK / 64), 128, 0, stream>>>(
            e_bf16, 128, wih0p + (size_t)d * 512 * 128, 128,
            gx0 + (size_t)d * TOK * GATES, GATES, GATES, 128);

    // 4) layer-0 recurrence
    lstm_dir_kernel<<<2, 512, lstm_smem, stream>>>(gx0, Whh0, bih0, bhh0, h0cat);

    // 5) layer-1 input GEMMs: gx1[d] = h0cat @ Wih1[d].T (K padded 250->256)
    for (int d = 0; d < 2; ++d)
        gemm_bf16_wmma<<<dim3(32, TOK / 64), 128, 0, stream>>>(
            h0cat, 256, wih1p + (size_t)d * 512 * 256, 256,
            gx1 + (size_t)d * TOK * GATES, GATES, GATES, 256);

    // 6) layer-1 recurrence
    lstm_dir_kernel<<<2, 512, lstm_smem, stream>>>(gx1, Whh1, bih1, bhh1, h1cat);

    // 7) projections: ph = h1 @ Wh.T ; pm = h1 @ Wm.T
    gemm_bf16_wmma<<<dim3(7, TOK / 64), 128, 0, stream>>>(h1cat, 256, whp, 256, ph, PROJ, PROJ, 256);
    gemm_bf16_wmma<<<dim3(7, TOK / 64), 128, 0, stream>>>(h1cat, 256, wmp, 256, pm, PROJ, PROJ, 256);

    // 8) fused tanh scoring -> sT (j-major, coalesced in i)
    score_kernel<<<dim3(TOK / 32, TOK / 32), dim3(16, 16), 0, stream>>>(ph, pm, b1, W2, b2, sT);

    // 9) column log_softmax -> out
    colsoftmax_kernel<<<TOK, 256, 0, stream>>>(sT, out);
}
